// RNN_91070486544568
// MI455X (gfx1250) — compile-verified, matrix operand order checked by reading, not run on hardware
//
#include <hip/hip_runtime.h>
#include <hip/hip_bf16.h>
#include <math.h>

// Problem dimensions (compile-time, match the reference).
#define T_    512
#define B_    1024
#define DIM_  128
#define LAT_  256
#define OUT_  64

typedef __attribute__((ext_vector_type(16))) __bf16 v16bf;
typedef __attribute__((ext_vector_type(8)))  float  v8f;

// float -> bf16 bits, round-to-nearest-even (h/weight writes).
__device__ __forceinline__ unsigned short f2bf(float f) {
    unsigned int b = __builtin_bit_cast(unsigned int, f);
    b += 0x7FFFu + ((b >> 16) & 1u);
    return (unsigned short)(b >> 16);
}

union FragBF {
    v16bf v;
    unsigned int p[8];
    uint4 q[2];
};

// Pack two f32 bit-patterns into one DWORD of two bf16 (truncation) with a
// single v_perm_b32: result bytes = {b0[2], b0[3], b1[2], b1[3]}.
__device__ __forceinline__ unsigned int pack_bf16_trunc(unsigned int b0, unsigned int b1) {
    return __builtin_amdgcn_perm(b1, b0, 0x07060302u);
}

// 16-element bf16 fragment slice from an f32 row (truncating convert).
// Per ISA 7.12.2 (16-bit A 16x32): lane holds K = {s..s+7, s+16..s+23}.
__device__ __forceinline__ v16bf load_frag_f32(const float* __restrict__ rowp, int s) {
    uint4 f0 = *(const uint4*)(rowp + s);
    uint4 f1 = *(const uint4*)(rowp + s + 4);
    uint4 f2 = *(const uint4*)(rowp + s + 16);
    uint4 f3 = *(const uint4*)(rowp + s + 20);
    FragBF fr;
    fr.p[0] = pack_bf16_trunc(f0.x, f0.y);
    fr.p[1] = pack_bf16_trunc(f0.z, f0.w);
    fr.p[2] = pack_bf16_trunc(f1.x, f1.y);
    fr.p[3] = pack_bf16_trunc(f1.z, f1.w);
    fr.p[4] = pack_bf16_trunc(f2.x, f2.y);
    fr.p[5] = pack_bf16_trunc(f2.z, f2.w);
    fr.p[6] = pack_bf16_trunc(f3.x, f3.y);
    fr.p[7] = pack_bf16_trunc(f3.z, f3.w);
    return fr.v;
}

// 16-element bf16 fragment slice from a bf16 row (2x global_load_b128).
__device__ __forceinline__ v16bf load_frag_bf16(const unsigned short* __restrict__ rowp, int s) {
    FragBF fr;
    fr.q[0] = *(const uint4*)(rowp + s);
    fr.q[1] = *(const uint4*)(rowp + s + 16);
    return fr.v;
}

// Convert W[K][N] (f32, row-major) into Wt[N][K] (bf16 bits, RNE).
__global__ void conv_transpose_kernel(const float* __restrict__ W,
                                      unsigned short* __restrict__ Wt,
                                      int K, int N) {
    int idx = blockIdx.x * blockDim.x + threadIdx.x;
    if (idx >= K * N) return;
    int k = idx / N;
    int n = idx - k * N;
    Wt[(size_t)n * K + k] = f2bf(W[(size_t)k * N + n]);
}

// Convert h0 (f32) -> bf16 working buffer (flat copy, RNE).
__global__ void conv_h_kernel(const float* __restrict__ h0,
                              unsigned short* __restrict__ hb) {
    int idx = blockIdx.x * blockDim.x + threadIdx.x;
    if (idx < B_ * LAT_) hb[idx] = f2bf(h0[idx]);
}

// One recurrence step: h_out = tanh( x_t @ Wi + bi + h_in @ Wh ).
// Register-blocked: each wave computes a 16(M) x 64(N) strip = 4 tiles, so A
// fragments (x rows, h rows) and their conversions are amortized 4x.
// Grid: 32 blocks x 256 threads = 256 waves; EXEC all-ones throughout.
__global__ void __launch_bounds__(256)
rnn_step_kernel(const float* __restrict__ xt,              // [B_, DIM_] f32
                const float* __restrict__ xnext,           // next step's x (prefetch)
                const unsigned short* __restrict__ hin,    // [B_, LAT_] bf16
                const unsigned short* __restrict__ WiT,    // [LAT_][DIM_] bf16
                const unsigned short* __restrict__ WhT,    // [LAT_][LAT_] bf16
                const float* __restrict__ bi,              // [LAT_]
                unsigned short* __restrict__ hout)         // [B_, LAT_] bf16
{
    const int lane   = threadIdx.x & 31;
    const int wave   = threadIdx.x >> 5;
    const int strip  = blockIdx.x * 8 + wave;   // 0..255
    const int nGroup = strip & 3;               // 4 groups of 4 n-tiles
    const int mTile  = strip >> 2;              // 64 m-tiles
    const int laneLo = lane & 15;
    const int hiHalf = lane >> 4;
    const int kb     = hiHalf * 8;              // per-lane K sub-offset

    const int row = mTile * 16 + laneLo;        // A-matrix row (batch index)

    int col[4];
    v8f acc[4];
#pragma unroll
    for (int j = 0; j < 4; ++j) {
        col[j] = (nGroup * 4 + j) * 16 + laneLo;
        const float bias = bi[col[j]];
        acc[j] = (v8f){bias, bias, bias, bias, bias, bias, bias, bias};
    }

    // Prefetch next step's x rows into cache while this step computes
    // (the only per-step HBM traffic; h and weights are L2-resident).
    {
        const char* pf = (const char*)(xnext + (size_t)row * DIM_) + hiHalf * 256;
        __builtin_prefetch(pf, 0, 3);
        __builtin_prefetch(pf + 128, 0, 3);
    }

    // acc[j] += x_t @ Wi   (K = 128 -> 4 chunks x 4 tiles = 16 WMMAs)
    const float* xrow = xt + (size_t)row * DIM_;
#pragma unroll
    for (int k0 = 0; k0 < DIM_; k0 += 32) {
        v16bf a = load_frag_f32(xrow, k0 + kb);
#pragma unroll
        for (int j = 0; j < 4; ++j) {
            v16bf b = load_frag_bf16(WiT + (size_t)col[j] * DIM_, k0 + kb);
            acc[j] = __builtin_amdgcn_wmma_f32_16x16x32_bf16(false, a, false, b,
                                                             (short)0, acc[j], false, false);
        }
    }

    // acc[j] += h_in @ Wh  (K = 256 -> 8 chunks x 4 tiles = 32 WMMAs)
    const unsigned short* hrow = hin + (size_t)row * LAT_;
#pragma unroll
    for (int k0 = 0; k0 < LAT_; k0 += 32) {
        v16bf a = load_frag_bf16(hrow, k0 + kb);
#pragma unroll
        for (int j = 0; j < 4; ++j) {
            v16bf b = load_frag_bf16(WhT + (size_t)col[j] * LAT_, k0 + kb);
            acc[j] = __builtin_amdgcn_wmma_f32_16x16x32_bf16(false, a, false, b,
                                                             (short)0, acc[j], false, false);
        }
    }

    // C/D layout: VGPR r -> M = mTile*16 + hiHalf*8 + r, N = col[j].
    const int mBase = mTile * 16 + hiHalf * 8;
#pragma unroll
    for (int j = 0; j < 4; ++j) {
#pragma unroll
        for (int r = 0; r < 8; ++r) {
            hout[(size_t)(mBase + r) * LAT_ + col[j]] = f2bf(tanhf(acc[j][r]));
        }
    }
}

// Head: out = sigmoid( h @ Wd + bd ).  64 x 4 = 256 tiles -> 32 blocks x 8 waves.
__global__ void __launch_bounds__(256)
rnn_head_kernel(const unsigned short* __restrict__ hin,    // [B_, LAT_] bf16
                const unsigned short* __restrict__ WdT,    // [OUT_][LAT_] bf16
                const float* __restrict__ bd,              // [OUT_]
                float* __restrict__ out)                   // [B_, OUT_] f32
{
    const int lane   = threadIdx.x & 31;
    const int wave   = threadIdx.x >> 5;
    const int tile   = blockIdx.x * 8 + wave;   // 0..255
    const int nTile  = tile & 3;                // OUT_/16 = 4 tiles
    const int mTile  = tile >> 2;               // B_/16  = 64 tiles
    const int laneLo = lane & 15;
    const int hiHalf = lane >> 4;
    const int kb     = hiHalf * 8;

    const int row = mTile * 16 + laneLo;
    const int col = nTile * 16 + laneLo;

    const float bias = bd[col];
    v8f acc = {bias, bias, bias, bias, bias, bias, bias, bias};

    const unsigned short* hrow  = hin + (size_t)row * LAT_;
    const unsigned short* wdcol = WdT + (size_t)col * LAT_;
#pragma unroll
    for (int k0 = 0; k0 < LAT_; k0 += 32) {
        v16bf a = load_frag_bf16(hrow, k0 + kb);
        v16bf b = load_frag_bf16(wdcol, k0 + kb);
        acc = __builtin_amdgcn_wmma_f32_16x16x32_bf16(false, a, false, b,
                                                      (short)0, acc, false, false);
    }

    const int mBase = mTile * 16 + hiHalf * 8;
#pragma unroll
    for (int r = 0; r < 8; ++r) {
        out[(size_t)(mBase + r) * OUT_ + col] = 1.0f / (1.0f + expf(-acc[r]));
    }
}

extern "C" void kernel_launch(void* const* d_in, const int* in_sizes, int n_in,
                              void* d_out, int out_size, void* d_ws, size_t ws_size,
                              hipStream_t stream) {
    (void)in_sizes; (void)n_in; (void)out_size; (void)ws_size;

    const float* x  = (const float*)d_in[0];   // [T_, B_, DIM_]
    const float* h0 = (const float*)d_in[1];   // [B_, LAT_] (zeros)
    const float* Wi = (const float*)d_in[2];   // [DIM_, LAT_]
    const float* bi = (const float*)d_in[3];   // [LAT_]
    const float* Wh = (const float*)d_in[4];   // [LAT_, LAT_]
    const float* Wd = (const float*)d_in[5];   // [LAT_, OUT_]
    const float* bd = (const float*)d_in[6];   // [OUT_]
    float* out = (float*)d_out;                // [B_, OUT_]

    // Workspace layout (~1.25 MB total):
    char* ws = (char*)d_ws;
    unsigned short* WiT = (unsigned short*)(ws);               //  64 KB: [LAT_][DIM_]
    unsigned short* WhT = (unsigned short*)(ws + (64 << 10));  // 128 KB: [LAT_][LAT_]
    unsigned short* WdT = (unsigned short*)(ws + (192 << 10)); //  32 KB: [OUT_][LAT_]
    unsigned short* hA  = (unsigned short*)(ws + (224 << 10)); // 512 KB: [B_][LAT_] bf16
    unsigned short* hB  = hA + (size_t)B_ * LAT_;              // 512 KB: [B_][LAT_] bf16

    // One-time prep: weights transposed+converted (L2-resident), h0 -> bf16.
    conv_transpose_kernel<<<(DIM_ * LAT_ + 255) / 256, 256, 0, stream>>>(Wi, WiT, DIM_, LAT_);
    conv_transpose_kernel<<<(LAT_ * LAT_ + 255) / 256, 256, 0, stream>>>(Wh, WhT, LAT_, LAT_);
    conv_transpose_kernel<<<(LAT_ * OUT_ + 255) / 256, 256, 0, stream>>>(Wd, WdT, LAT_, OUT_);
    conv_h_kernel<<<(B_ * LAT_ + 255) / 256, 256, 0, stream>>>(h0, hA);

    // Serial recurrence: 512 dependent launches, ping-ponging bf16 h buffers.
    const unsigned short* cur = hA;
    for (int t = 0; t < T_; ++t) {
        unsigned short* nxt = (cur == hA) ? hB : hA;
        const float* xt = x + (size_t)t * B_ * DIM_;
        const float* xn = (t + 1 < T_) ? (xt + (size_t)B_ * DIM_) : xt;
        rnn_step_kernel<<<32, 256, 0, stream>>>(xt, xn, cur, WiT, WhT, bi, nxt);
        cur = nxt;
    }

    // Head projection + sigmoid.
    rnn_head_kernel<<<32, 256, 0, stream>>>(cur, WdT, bd, out);
}
